// HierarchicalMoEMLPBlock_24842090840238
// MI455X (gfx1250) — compile-verified, matrix operand order checked by reading, not run on hardware
//
#include <hip/hip_runtime.h>
#include <hip/hip_bf16.h>
#include <math.h>

// Problem constants (match reference)
#define GNUM 4
#define ENUM 8
#define NTOK 512
#define DDIM 768
#define MDIM 3072
#define NEXP 32      // GNUM*ENUM
#define CAP  512     // max tokens per expert
#define HROWS 2048   // NTOK * (KG*KE) assignment rows in h

typedef __attribute__((ext_vector_type(16))) __bf16 v16bf;
typedef __attribute__((ext_vector_type(8)))  __bf16 v8bf;
typedef __attribute__((ext_vector_type(4)))  __bf16 v4bf;
typedef __attribute__((ext_vector_type(8)))  float  v8f;
typedef __attribute__((ext_vector_type(4)))  float  v4f;

__device__ __forceinline__ float gelu_exact(float v) {
    return 0.5f * v * (1.0f + erff(v * 0.70710678118654752f));
}

// ---------------------------------------------------------------------------
// Kernel 1: two-level top-k routing. One thread per token.
// ---------------------------------------------------------------------------
__global__ void route_kernel(const float* __restrict__ x,
                             const float* __restrict__ Wg,   // [D][G]
                             const float* __restrict__ bg,   // [G]
                             const float* __restrict__ Wge,  // [G][D][E]
                             const float* __restrict__ bge,  // [G][E]
                             int*   __restrict__ cnt,        // [NEXP]
                             int*   __restrict__ tok_list,   // [NEXP][CAP]
                             float* __restrict__ wgt_list)   // [NEXP][CAP]
{
    int n = blockIdx.x * blockDim.x + threadIdx.x;
    if (n >= NTOK) return;
    const float* xr = x + (size_t)n * DDIM;

    float gl[GNUM];
    #pragma unroll
    for (int g = 0; g < GNUM; ++g) gl[g] = bg[g];
    for (int k = 0; k < DDIM; ++k) {
        float xv = xr[k];
        #pragma unroll
        for (int g = 0; g < GNUM; ++g) gl[g] = fmaf(xv, Wg[k * GNUM + g], gl[g]);
    }
    int g0 = 0;
    #pragma unroll
    for (int g = 1; g < GNUM; ++g) if (gl[g] > gl[g0]) g0 = g;
    int g1 = -1;
    #pragma unroll
    for (int g = 0; g < GNUM; ++g) { if (g == g0) continue; if (g1 < 0 || gl[g] > gl[g1]) g1 = g; }
    float geb = __expf(gl[g1] - gl[g0]);
    float ggate[2] = { 1.0f / (1.0f + geb), geb / (1.0f + geb) };
    int   gsel[2]  = { g0, g1 };

    float el[2][ENUM];
    #pragma unroll
    for (int i = 0; i < 2; ++i)
        #pragma unroll
        for (int e = 0; e < ENUM; ++e) el[i][e] = bge[gsel[i] * ENUM + e];
    for (int k = 0; k < DDIM; ++k) {
        float xv = xr[k];
        #pragma unroll
        for (int i = 0; i < 2; ++i) {
            const float* wp = Wge + ((size_t)gsel[i] * DDIM + k) * ENUM;
            #pragma unroll
            for (int e = 0; e < ENUM; ++e) el[i][e] = fmaf(xv, wp[e], el[i][e]);
        }
    }
    #pragma unroll
    for (int i = 0; i < 2; ++i) {
        int e0 = 0;
        #pragma unroll
        for (int e = 1; e < ENUM; ++e) if (el[i][e] > el[i][e0]) e0 = e;
        int e1 = -1;
        #pragma unroll
        for (int e = 0; e < ENUM; ++e) { if (e == e0) continue; if (e1 < 0 || el[i][e] > el[i][e1]) e1 = e; }
        float eeb = __expf(el[i][e1] - el[i][e0]);
        float egs[2] = { 1.0f / (1.0f + eeb), eeb / (1.0f + eeb) };
        int   eis[2] = { e0, e1 };
        #pragma unroll
        for (int j = 0; j < 2; ++j) {
            int   eid = gsel[i] * ENUM + eis[j];
            float w   = ggate[i] * egs[j];
            int pos = atomicAdd(&cnt[eid], 1);
            tok_list[eid * CAP + pos] = n;
            wgt_list[eid * CAP + pos] = w;
        }
    }
}

// ---------------------------------------------------------------------------
// Kernel 2: exclusive scan of per-expert counts (32 entries).
// ---------------------------------------------------------------------------
__global__ void scan_kernel(const int* __restrict__ cnt, int* __restrict__ base)
{
    if (threadIdx.x == 0) {
        int run = 0;
        for (int e = 0; e < NEXP; ++e) { base[e] = run; run += cnt[e]; }
    }
}

// ---------------------------------------------------------------------------
// Kernel 3: fc1 + GELU.  h[base[e]+r][m] = gelu(x[tok] @ W1[e] + b1[e])
// 256 threads (8 waves, 2x4 wave grid), tile 32(n) x 64(m), K-step 32 (bf16),
// double-buffered LDS, one barrier per K-step, one WMMA per wave per K-step.
// ---------------------------------------------------------------------------
__global__ void __launch_bounds__(256)
fc1_kernel(const float* __restrict__ x,
           const float* __restrict__ W1,        // [NEXP][D][M]
           const float* __restrict__ b1,        // [NEXP][M]
           const int*   __restrict__ cnt,
           const int*   __restrict__ base,
           const int*   __restrict__ tok_list,
           float*       __restrict__ h)         // [HROWS][M]
{
    const int e  = blockIdx.z;
    const int n0 = blockIdx.y * 32;
    const int m0 = blockIdx.x * 64;
    const int ne = cnt[e];
    if (n0 >= ne) return;

    __shared__ __align__(32) __bf16 xs[2][32][32];    // A tile: rows x K
    __shared__ __align__(32) __bf16 wsb[2][64][32];   // B tile transposed: cols x K
    __shared__ int tok_s[32];

    const int t    = threadIdx.x;
    const int lane = t & 31;
    const int wave = t >> 5;
    const int wn   = wave >> 2;   // 0..1 n sub-tile
    const int wm   = wave & 3;    // 0..3 m sub-tile

    if (t < 32) tok_s[t] = (n0 + t < ne) ? tok_list[e * CAP + n0 + t] : 0;
    __syncthreads();

    // --- per-thread staging geometry (loop-invariant) ---
    const int  ar    = t >> 3;          // A row 0..31
    const int  aks   = (t & 7) * 4;     // A k-segment start
    const float aflag = (n0 + ar < ne) ? 1.0f : 0.0f;
    const float* aptr = x + (size_t)tok_s[ar] * DDIM + aks;   // + k0 each step

    const int  wcol  = t & 63;          // W output column (m)
    const int  wk8   = (t >> 6) * 8;    // W k-segment start (8 deep)
    const float* W1e  = W1 + (size_t)e * DDIM * MDIM;
    const float* wptr = W1e + (size_t)wk8 * MDIM + m0 + wcol; // + (k0+j)*M each step

    // --- fragment geometry ---
    const int ak   = (lane >> 4) * 16;
    const int arow = wn * 16 + (lane & 15);
    const int brow = wm * 16 + (lane & 15);

    float areg[4], wreg[8];
    // preload k0 = 0
    {
        v4f av = *(const v4f*)aptr;
        #pragma unroll
        for (int j = 0; j < 4; ++j) areg[j] = av[j] * aflag;
        #pragma unroll
        for (int j = 0; j < 8; ++j) wreg[j] = wptr[(size_t)j * MDIM];
    }

    v8f c = {};
    int cur = 0;
    for (int k0 = 0; k0 < DDIM; k0 += 32) {
        // store staged data (packed bf16, vector LDS stores)
        v4bf ap;
        #pragma unroll
        for (int j = 0; j < 4; ++j) ap[j] = (__bf16)areg[j];
        *(v4bf*)&xs[cur][ar][aks] = ap;
        v8bf wp;
        #pragma unroll
        for (int j = 0; j < 8; ++j) wp[j] = (__bf16)wreg[j];
        *(v8bf*)&wsb[cur][wcol][wk8] = wp;

        // issue next iteration's global loads before the barrier
        if (k0 + 32 < DDIM) {
            v4f av = *(const v4f*)(aptr + k0 + 32);
            #pragma unroll
            for (int j = 0; j < 4; ++j) areg[j] = av[j] * aflag;
            #pragma unroll
            for (int j = 0; j < 8; ++j) wreg[j] = wptr[(size_t)(k0 + 32 + j) * MDIM];
            if (k0 + 64 < DDIM)
                __builtin_prefetch(wptr + (size_t)(k0 + 64) * MDIM, 0, 1);
        }
        __syncthreads();

        v16bf a = *(const v16bf*)&xs[cur][arow][ak];
        v16bf b = *(const v16bf*)&wsb[cur][brow][ak];
        c = __builtin_amdgcn_wmma_f32_16x16x32_bf16(false, a, false, b,
                                                    (short)0, c, false, false);
        cur ^= 1;
    }

    const int   colm = m0 + wm * 16 + (lane & 15);
    const float bb   = b1[(size_t)e * MDIM + colm];
    const int   hi   = (lane >> 4) * 8;      // C layout: VGPR i -> row i / i+8
    const size_t hb  = (size_t)(base[e] + n0) * MDIM;
    #pragma unroll
    for (int i = 0; i < 8; ++i) {
        int lr = wn * 16 + i + hi;
        if (n0 + lr < ne)
            h[hb + (size_t)lr * MDIM + colm] = gelu_exact(c[i] + bb);
    }
}

// ---------------------------------------------------------------------------
// Kernel 4: fc2 + weighted scatter. out[tok] += w * (h @ W2[e] + b2[e])
// ---------------------------------------------------------------------------
__global__ void __launch_bounds__(256)
fc2_kernel(const float* __restrict__ h,         // [HROWS][M]
           const float* __restrict__ W2,        // [NEXP][M][D]
           const float* __restrict__ b2,        // [NEXP][D]
           const int*   __restrict__ cnt,
           const int*   __restrict__ base,
           const int*   __restrict__ tok_list,
           const float* __restrict__ wgt_list,
           float*       __restrict__ out)       // [NTOK][D]
{
    const int e  = blockIdx.z;
    const int n0 = blockIdx.y * 32;
    const int d0 = blockIdx.x * 64;
    const int ne = cnt[e];
    if (n0 >= ne) return;

    __shared__ __align__(32) __bf16 hsb[2][32][32];
    __shared__ __align__(32) __bf16 wsb[2][64][32];
    __shared__ int   tok_s[32];
    __shared__ float wgt_s[32];

    const int t    = threadIdx.x;
    const int lane = t & 31;
    const int wave = t >> 5;
    const int wn   = wave >> 2;
    const int wm   = wave & 3;

    if (t < 32) {
        bool ok = (n0 + t < ne);
        tok_s[t] = ok ? tok_list[e * CAP + n0 + t] : 0;
        wgt_s[t] = ok ? wgt_list[e * CAP + n0 + t] : 0.0f;
    }
    __syncthreads();

    const int  ar    = t >> 3;
    const int  aks   = (t & 7) * 4;
    const float aflag = (n0 + ar < ne) ? 1.0f : 0.0f;
    int hrow = base[e] + n0 + ar;
    if (hrow > HROWS - 1) hrow = HROWS - 1;          // clamp: stay in h buffer
    const float* aptr = h + (size_t)hrow * MDIM + aks;

    const int  wcol  = t & 63;
    const int  wk8   = (t >> 6) * 8;
    const float* W2e  = W2 + (size_t)e * MDIM * DDIM;
    const float* wptr = W2e + (size_t)wk8 * DDIM + d0 + wcol;

    const int ak   = (lane >> 4) * 16;
    const int arow = wn * 16 + (lane & 15);
    const int brow = wm * 16 + (lane & 15);

    float areg[4], wreg[8];
    {
        v4f av = *(const v4f*)aptr;
        #pragma unroll
        for (int j = 0; j < 4; ++j) areg[j] = av[j] * aflag;
        #pragma unroll
        for (int j = 0; j < 8; ++j) wreg[j] = wptr[(size_t)j * DDIM];
    }

    v8f c = {};
    int cur = 0;
    for (int k0 = 0; k0 < MDIM; k0 += 32) {
        v4bf ap;
        #pragma unroll
        for (int j = 0; j < 4; ++j) ap[j] = (__bf16)areg[j];
        *(v4bf*)&hsb[cur][ar][aks] = ap;
        v8bf wp;
        #pragma unroll
        for (int j = 0; j < 8; ++j) wp[j] = (__bf16)wreg[j];
        *(v8bf*)&wsb[cur][wcol][wk8] = wp;

        if (k0 + 32 < MDIM) {
            v4f av = *(const v4f*)(aptr + k0 + 32);
            #pragma unroll
            for (int j = 0; j < 4; ++j) areg[j] = av[j] * aflag;
            #pragma unroll
            for (int j = 0; j < 8; ++j) wreg[j] = wptr[(size_t)(k0 + 32 + j) * DDIM];
            if (k0 + 64 < MDIM)
                __builtin_prefetch(wptr + (size_t)(k0 + 64) * DDIM, 0, 1);
        }
        __syncthreads();

        v16bf a = *(const v16bf*)&hsb[cur][arow][ak];
        v16bf b = *(const v16bf*)&wsb[cur][brow][ak];
        c = __builtin_amdgcn_wmma_f32_16x16x32_bf16(false, a, false, b,
                                                    (short)0, c, false, false);
        cur ^= 1;
    }

    const int   cold = d0 + wm * 16 + (lane & 15);
    const float bb   = b2[(size_t)e * DDIM + cold];
    const int   hi   = (lane >> 4) * 8;
    #pragma unroll
    for (int i = 0; i < 8; ++i) {
        int lr = wn * 16 + i + hi;
        if (n0 + lr < ne) {
            float v = (c[i] + bb) * wgt_s[lr];
            atomicAdd(&out[(size_t)tok_s[lr] * DDIM + cold], v);
        }
    }
}

// ---------------------------------------------------------------------------
// Host launcher
// ---------------------------------------------------------------------------
extern "C" void kernel_launch(void* const* d_in, const int* in_sizes, int n_in,
                              void* d_out, int out_size, void* d_ws, size_t ws_size,
                              hipStream_t stream)
{
    const float* x   = (const float*)d_in[0];
    const float* Wg  = (const float*)d_in[1];
    const float* bg  = (const float*)d_in[2];
    const float* Wge = (const float*)d_in[3];
    const float* bge = (const float*)d_in[4];
    const float* W1  = (const float*)d_in[5];
    const float* b1  = (const float*)d_in[6];
    const float* W2  = (const float*)d_in[7];
    const float* b2  = (const float*)d_in[8];
    float* out = (float*)d_out;

    // Workspace layout (128B aligned):
    //   [0,128)                  cnt[32]
    //   [128,256)                base[32]
    //   [256, +64K)              tok_list[32][512]
    //   next 64K                 wgt_list[32][512]
    //   next 24M                 h[2048][3072]
    char*  ws       = (char*)d_ws;
    int*   cnt      = (int*)(ws);
    int*   base     = (int*)(ws + 128);
    int*   tok_list = (int*)(ws + 256);
    float* wgt_list = (float*)(ws + 256 + (size_t)NEXP * CAP * 4);
    float* h        = (float*)(ws + 256 + (size_t)2 * NEXP * CAP * 4);

    hipMemsetAsync(cnt, 0, NEXP * sizeof(int), stream);
    hipMemsetAsync(out, 0, (size_t)NTOK * DDIM * sizeof(float), stream);

    route_kernel<<<NTOK / 128, 128, 0, stream>>>(x, Wg, bg, Wge, bge,
                                                 cnt, tok_list, wgt_list);
    scan_kernel<<<1, 32, 0, stream>>>(cnt, base);

    dim3 g1(MDIM / 64, NTOK / 32, NEXP);
    fc1_kernel<<<g1, 256, 0, stream>>>(x, W1, b1, cnt, base, tok_list, h);

    dim3 g2(DDIM / 64, NTOK / 32, NEXP);
    fc2_kernel<<<g2, 256, 0, stream>>>(h, W2, b2, cnt, base, tok_list, wgt_list, out);
}
